// KimiDeltaAttention_24773371363692
// MI455X (gfx1250) — compile-verified, matrix operand order checked by reading, not run on hardware
//
#include <hip/hip_runtime.h>

// ---------------------------------------------------------------------------
// Kimi Delta Attention forward — CDNA5 (gfx1250), wave32.
// WMMA bf16 everywhere (projections + chunked delta-rule), TDM async tile
// load, vectorized ds_load_b128 fragment gathers.
// ---------------------------------------------------------------------------

typedef __attribute__((ext_vector_type(16))) __bf16 v16bf;
typedef __attribute__((ext_vector_type(8), may_alias)) __bf16 v8bf;
typedef __attribute__((ext_vector_type(8)))  float  v8f;
typedef __attribute__((ext_vector_type(4))) unsigned int u32x4;
typedef __attribute__((ext_vector_type(8))) int i32x8;
typedef __attribute__((ext_vector_type(4))) int i32x4;

// Problem constants (match reference)
constexpr int Bc      = 2;
constexpr int Tc      = 4096;
constexpr int HID     = 2048;
constexpr int NH      = 16;
constexpr int DKc     = 128;
constexpr int DVc     = 128;
constexpr int KEY_DIM = NH * DKc;   // 2048
constexpr int VAL_DIM = NH * DVc;   // 2048
constexpr int CONV_K  = 4;
constexpr int CHUNK   = 64;
constexpr int NCHUNK  = Tc / CHUNK; // 64
constexpr int Mrows   = Bc * Tc;    // 8192
constexpr float EPSf  = 1e-5f;

constexpr int DVS    = 32;          // dv slice per attention block
constexpr int NSPLIT = DVc / DVS;   // 4

// ---------------------------------------------------------------------------
// bf16 helpers
// ---------------------------------------------------------------------------
__device__ __forceinline__ unsigned short f2bf(float f) {
  unsigned u = __float_as_uint(f);
  unsigned r = ((u >> 16) & 1u) + 0x7fffu;    // round-to-nearest-even
  return (unsigned short)((u + r) >> 16);
}
__device__ __forceinline__ float bf2f(unsigned short h) {
  return __uint_as_float(((unsigned)h) << 16);
}

// Fragment load: 16 bf16 elements for one WMMA operand lane-row, as two
// aligned 16B LDS loads (ds_load_b128).  Per ISA 7.12.2 the element->K map is
//   e in [0,8):  K = k0 + laneHi*8 + e
//   e in [8,16): K = k0 + 16 + laneHi*8 + (e-8)
// valid when the operand tile is stored row-major with the contraction dim
// contiguous (row = M for A, row = N for "transposed-B" storage).
__device__ __forceinline__ v16bf frag_ld(const unsigned short* base, int row,
                                         int stride, int k0, int laneHi) {
  const unsigned short* p = base + (long)row * stride + k0 + laneHi * 8;
  v8bf lo = *reinterpret_cast<const v8bf*>(p);
  v8bf hi = *reinterpret_cast<const v8bf*>(p + 16);
  return __builtin_shufflevector(lo, hi, 0, 1, 2, 3, 4, 5, 6, 7, 8, 9, 10, 11,
                                 12, 13, 14, 15);
}

// Accumulate one 16x16 tile over KD (both operands LDS, contraction-contig).
template <int KD>
__device__ __forceinline__ v8f wmma_acc(const unsigned short* A, int rowA,
                                        int sA, const unsigned short* B,
                                        int colB, int sB, int laneHi,
                                        int lanelo, v8f acc) {
#pragma unroll
  for (int k0 = 0; k0 < KD; k0 += 32) {
    v16bf a = frag_ld(A, rowA + lanelo, sA, k0, laneHi);
    v16bf b = frag_ld(B, colB + lanelo, sB, k0, laneHi);
    acc = __builtin_amdgcn_wmma_f32_16x16x32_bf16(false, a, false, b, (short)0,
                                                  acc, false, false);
  }
  return acc;
}

__device__ __forceinline__ v8f v8f_zero() {
  v8f z;
#pragma unroll
  for (int e = 0; e < 8; e++) z[e] = 0.f;
  return z;
}

// ---------------------------------------------------------------------------
// Tensor Data Mover: async 2D f32 tile (tileW x tileH elems, row stride in
// elems) from global to LDS byte offset lds_off.  D# packed per ISA 8.3/8.4.
// ---------------------------------------------------------------------------
__device__ __forceinline__ void tdm_load_2d_f32(unsigned lds_off,
                                                unsigned long long gaddr,
                                                unsigned tileW, unsigned tileH,
                                                unsigned long long strideElems) {
  u32x4 g0;
  g0[0] = 1u;                                            // count=1 (user D#)
  g0[1] = lds_off;                                       // lds_addr (bytes)
  g0[2] = (unsigned)(gaddr & 0xffffffffu);               // global_addr lo
  g0[3] = (unsigned)((gaddr >> 32) & 0x1ffffffu) | (2u << 30);  // hi | type=2
  i32x8 g1;
  g1[0] = (int)(2u << 16);                     // wg_mask=0, data_size=4B
  g1[1] = (int)(tileW << 16);                  // tensor_dim0[15:0]
  g1[2] = (int)((tileH & 0xffffu) << 16);      // tensor_dim0 hi=0 | dim1 lo
  g1[3] = (int)((tileW & 0xffffu) << 16);      // dim1 hi=0 | tile_dim0
  g1[4] = (int)(tileH & 0xffffu);              // tile_dim1 (tile_dim2=0)
  g1[5] = (int)(unsigned)(strideElems & 0xffffffffu);   // dim0_stride lo
  g1[6] = (int)(unsigned)((strideElems >> 32) & 0xffffu);
  g1[7] = 0;
  i32x4 z4;
  z4[0] = 0; z4[1] = 0; z4[2] = 0; z4[3] = 0;
#if __clang_major__ >= 23
  i32x8 z8;
#pragma unroll
  for (int e = 0; e < 8; e++) z8[e] = 0;
  __builtin_amdgcn_tensor_load_to_lds(g0, g1, z4, z4, z8, 0);
#else
  __builtin_amdgcn_tensor_load_to_lds(g0, g1, z4, z4, 0);
#endif
}

// ---------------------------------------------------------------------------
// f32 -> bf16 cast (row-major, used for A-operand activations)
// ---------------------------------------------------------------------------
__global__ void cast_bf16_kernel(const float* __restrict__ src,
                                 unsigned short* __restrict__ dst, long n) {
  long stride = (long)gridDim.x * blockDim.x;
  for (long i = (long)blockIdx.x * blockDim.x + threadIdx.x; i < n; i += stride)
    dst[i] = f2bf(src[i]);
}

// ---------------------------------------------------------------------------
// Transpose-cast: src f32 [K x N] -> dst bf16 [N x K] (K contiguous), so the
// GEMM B operand can be staged and fragment-loaded fully vectorized.
// ---------------------------------------------------------------------------
__global__ __launch_bounds__(256) void castT_bf16_kernel(
    const float* __restrict__ src, unsigned short* __restrict__ dst, int Kd,
    int Nd) {
  __shared__ float tile[32][33];
  const int kb = blockIdx.x * 32, nb = blockIdx.y * 32;
  const int tx = threadIdx.x & 31, ty = threadIdx.x >> 5;  // 32x8
  for (int r = ty; r < 32; r += 8) {
    const int kk = kb + r, nn = nb + tx;
    tile[r][tx] = (kk < Kd && nn < Nd) ? src[(long)kk * Nd + nn] : 0.f;
  }
  __syncthreads();
  for (int r = ty; r < 32; r += 8) {
    const int nn = nb + r, kk = kb + tx;
    if (nn < Nd && kk < Kd) dst[(long)nn * Kd + kk] = f2bf(tile[tx][r]);
  }
}

// ---------------------------------------------------------------------------
// WMMA bf16 GEMM: C[f32, MxN] = A[bf16, MxK] * BT[bf16, NxK]^T
// 256 threads = 8 waves; block tile 128x128, K-step 32.
// Both LDS tiles stored contraction-contiguous, stride 40 (16B-aligned rows,
// conflict-free for b128 fragment loads).
// ---------------------------------------------------------------------------
__global__ __launch_bounds__(256) void gemm_bf16_wmma(
    const unsigned short* __restrict__ A, const unsigned short* __restrict__ BT,
    float* __restrict__ C, int Md, int Nd, int Kd) {
  constexpr int BM = 128, BN = 128, BK = 32;
  __shared__ __align__(16) unsigned short As[BM][BK + 8];
  __shared__ __align__(16) unsigned short BsT[BN][BK + 8];

  const int tid    = threadIdx.x;
  const int lane   = tid & 31;
  const int wave   = tid >> 5;
  const int waveM  = wave >> 1;   // 0..3
  const int waveN  = wave & 1;    // 0..1
  const int laneHi = lane >> 4;
  const int lanelo = lane & 15;
  const int blockM = blockIdx.y * BM;
  const int blockN = blockIdx.x * BN;

  v8f acc[2][4];
#pragma unroll
  for (int i = 0; i < 2; i++)
#pragma unroll
    for (int j = 0; j < 4; j++) acc[i][j] = v8f_zero();

  for (int k0 = 0; k0 < Kd; k0 += BK) {
    {  // stage A tile: 16 contiguous bf16 per thread, 2x b128
      const int r = tid >> 1;
      const int c = (tid & 1) * 16;
      const unsigned short* gp = &A[(long)(blockM + r) * Kd + k0 + c];
      uint4 p0 = *reinterpret_cast<const uint4*>(gp);
      uint4 p1 = *reinterpret_cast<const uint4*>(gp + 8);
      *reinterpret_cast<uint4*>(&As[r][c])     = p0;
      *reinterpret_cast<uint4*>(&As[r][c + 8]) = p1;
    }
    {  // stage BT tile (N-major, K contiguous): also fully vectorized
      const int cc = tid >> 1;
      const int kf = (tid & 1) * 16;
      if (blockN + cc < Nd) {
        const unsigned short* gp = &BT[(long)(blockN + cc) * Kd + k0 + kf];
        uint4 p0 = *reinterpret_cast<const uint4*>(gp);
        uint4 p1 = *reinterpret_cast<const uint4*>(gp + 8);
        *reinterpret_cast<uint4*>(&BsT[cc][kf])     = p0;
        *reinterpret_cast<uint4*>(&BsT[cc][kf + 8]) = p1;
      } else {
        uint4 z; z.x = z.y = z.z = z.w = 0u;
        *reinterpret_cast<uint4*>(&BsT[cc][kf])     = z;
        *reinterpret_cast<uint4*>(&BsT[cc][kf + 8]) = z;
      }
    }
    if (k0 + BK < Kd) {  // L2 prefetch of next K tiles (global_prefetch_b8)
      __builtin_prefetch(
          (const void*)&A[(long)(blockM + (tid >> 1)) * Kd + k0 + BK], 0, 1);
      if (blockN + (tid >> 1) < Nd)
        __builtin_prefetch(
            (const void*)&BT[(long)(blockN + (tid >> 1)) * Kd + k0 + BK], 0, 1);
    }
    __syncthreads();

    v16bf afrag[2], bfrag[4];
#pragma unroll
    for (int mt = 0; mt < 2; mt++)
      afrag[mt] = frag_ld(&As[0][0], waveM * 32 + mt * 16 + lanelo, BK + 8, 0,
                          laneHi);
#pragma unroll
    for (int nt = 0; nt < 4; nt++)
      bfrag[nt] = frag_ld(&BsT[0][0], waveN * 64 + nt * 16 + lanelo, BK + 8, 0,
                          laneHi);
#pragma unroll
    for (int mt = 0; mt < 2; mt++)
#pragma unroll
      for (int nt = 0; nt < 4; nt++)
        acc[mt][nt] = __builtin_amdgcn_wmma_f32_16x16x32_bf16(
            false, afrag[mt], false, bfrag[nt], (short)0, acc[mt][nt], false,
            false);
    __syncthreads();
  }

#pragma unroll
  for (int mt = 0; mt < 2; mt++) {
    const int rowb = blockM + waveM * 32 + mt * 16 + laneHi * 8;
#pragma unroll
    for (int nt = 0; nt < 4; nt++) {
      const int col = blockN + waveN * 64 + nt * 16 + lanelo;
      if (col < Nd) {
#pragma unroll
        for (int r = 0; r < 8; r++)
          C[(long)(rowb + r) * Nd + col] = acc[mt][nt][r];
      }
    }
  }
}

// ---------------------------------------------------------------------------
// Causal depthwise conv (K=4) + SiLU
// ---------------------------------------------------------------------------
__global__ void conv_silu_kernel(const float* __restrict__ pre,
                                 const float* __restrict__ kern,
                                 float* __restrict__ post, int C) {
  const long n = (long)Bc * Tc * C;
  const long stride = (long)gridDim.x * blockDim.x;
  for (long i = (long)blockIdx.x * blockDim.x + threadIdx.x; i < n; i += stride) {
    const int  c  = (int)(i % C);
    const long bt = i / C;
    const int  t  = (int)(bt % Tc);
    const long b  = bt / Tc;
    float acc = 0.f;
#pragma unroll
    for (int j = 0; j < CONV_K; j++) {
      const int tt = t - (CONV_K - 1) + j;
      if (tt >= 0) acc += pre[(b * Tc + tt) * C + c] * kern[j * C + c];
    }
    post[i] = acc / (1.f + __expf(-acc));
  }
}

// g = -exp(A_log[h]) * softplus(f + dt_bias[c])   (in place)
__global__ void f_to_g_kernel(float* __restrict__ f,
                              const float* __restrict__ A_log,
                              const float* __restrict__ dt_bias) {
  const long n = (long)Mrows * KEY_DIM;
  const long stride = (long)gridDim.x * blockDim.x;
  for (long i = (long)blockIdx.x * blockDim.x + threadIdx.x; i < n; i += stride) {
    const int c = (int)(i % KEY_DIM);
    const int h = c / DKc;
    const float x  = f[i] + dt_bias[c];
    const float sp = (x > 20.f) ? x : log1pf(__expf(x));
    f[i] = -__expf(A_log[h]) * sp;
  }
}

__global__ void sigmoid_kernel(float* __restrict__ p, long n) {
  const long stride = (long)gridDim.x * blockDim.x;
  for (long i = (long)blockIdx.x * blockDim.x + threadIdx.x; i < n; i += stride)
    p[i] = 1.f / (1.f + __expf(-p[i]));
}

// ---------------------------------------------------------------------------
// Chunked delta-rule scan, WMMA edition.  One block per (b, h, dv-slice).
// LDS layout (dynamic, ~180 KB; all offsets 16B aligned, strides chosen for
// conflict-free ds_load_b128):
//   Stf   f32 [32][128]  state S^T (dv-major)
//   St16  bf16[32][136]  bf16 shadow of S^T        (WMMA B operand)
//   kt    bf16[64][136]  k*e^g                      (A operand)
//   km    bf16[64][136]  k*e^{-g}                   (B^T operand)
//   qt    bf16[64][136]  q*e^g*DK^-1/2              (A operand)
//   kmT   bf16[128][72]  km transposed              (B^T operand for S update)
//   gcf   f32 [64][128]  cumsum(g)
//   Amat  f32 [64][64]   beta-scaled strict-tril system matrix
//   At16  bf16[64][72]   inclusive-tril attention   (A operand)
//   vntf  f32 [32][64]   v_new^T (solve output)
//   vnt16 bf16[32][72]   bf16 shadow                (A / B^T operand)
//   vc    f32 [64][32]   v slice (filled by TDM)
//   rks   f32 [64][32]   v - kt@S
//   glast f32 [128], betac f32 [64]
// ---------------------------------------------------------------------------
constexpr unsigned STF_OFF   = 0;
constexpr unsigned ST16_OFF  = STF_OFF   + 32 * 128 * 4;        // 16384
constexpr unsigned KT_OFF    = ST16_OFF  + 32 * 136 * 2;        // 25088
constexpr unsigned KM_OFF    = KT_OFF    + 64 * 136 * 2;        // 42496
constexpr unsigned QT_OFF    = KM_OFF    + 64 * 136 * 2;        // 59904
constexpr unsigned KMT_OFF   = QT_OFF    + 64 * 136 * 2;        // 77312
constexpr unsigned GCF_OFF   = KMT_OFF   + 128 * 72 * 2;        // 95744
constexpr unsigned AM_OFF    = GCF_OFF   + 64 * 128 * 4;        // 128512
constexpr unsigned AT16_OFF  = AM_OFF    + 64 * 64 * 4;         // 144896
constexpr unsigned VNT_OFF   = AT16_OFF  + 64 * 72 * 2;         // 154112
constexpr unsigned VNT16_OFF = VNT_OFF   + 32 * 64 * 4;         // 162304
constexpr unsigned VC_OFF    = VNT16_OFF + 32 * 72 * 2;         // 166912
constexpr unsigned RKS_OFF   = VC_OFF    + 64 * 32 * 4;         // 175104
constexpr unsigned GLAST_OFF = RKS_OFF   + 64 * 32 * 4;         // 183296
constexpr unsigned BETA_OFF  = GLAST_OFF + 128 * 4;             // 183808
constexpr unsigned ATT_SMEM  = BETA_OFF  + 64 * 4;              // 184064

__global__ __launch_bounds__(256) void kda_chunk_kernel(
    const float* __restrict__ q, const float* __restrict__ k,
    const float* __restrict__ v, const float* __restrict__ g,
    const float* __restrict__ beta, float* __restrict__ o) {
  extern __shared__ char smem[];
  float*          Stf   = (float*)(smem + STF_OFF);
  unsigned short* St16  = (unsigned short*)(smem + ST16_OFF);
  unsigned short* kt    = (unsigned short*)(smem + KT_OFF);
  unsigned short* km    = (unsigned short*)(smem + KM_OFF);
  unsigned short* qt    = (unsigned short*)(smem + QT_OFF);
  unsigned short* kmT   = (unsigned short*)(smem + KMT_OFF);
  float*          gcf   = (float*)(smem + GCF_OFF);
  float*          Amat  = (float*)(smem + AM_OFF);
  unsigned short* At16  = (unsigned short*)(smem + AT16_OFF);
  float*          vntf  = (float*)(smem + VNT_OFF);
  unsigned short* vnt16 = (unsigned short*)(smem + VNT16_OFF);
  float*          vc    = (float*)(smem + VC_OFF);
  float*          rks   = (float*)(smem + RKS_OFF);
  float*          glast = (float*)(smem + GLAST_OFF);
  float*          betac = (float*)(smem + BETA_OFF);

  const int tid    = threadIdx.x;
  const int lane   = tid & 31;
  const int wave   = tid >> 5;
  const int laneHi = lane >> 4;
  const int lanelo = lane & 15;
  const int vs  = blockIdx.x % NSPLIT;
  const int bh  = blockIdx.x / NSPLIT;
  const int h   = bh % NH;
  const int b   = bh / NH;
  const float qscale = 0.08838834764831845f;  // DK^-0.5

  for (int e = tid; e < 32 * 128; e += 256) Stf[e] = 0.f;
  for (int e = tid; e < 32 * 136; e += 256) St16[e] = 0;
  __syncthreads();

  for (int ci = 0; ci < NCHUNK; ci++) {
    const long t0 = (long)ci * CHUNK;

    // ---- Phase A: TDM v tile, cumsum(g), beta ----
    if (tid == 0) {
      const float* vp = v + (((long)b * Tc + t0) * NH + h) * DVc + vs * DVS;
      tdm_load_2d_f32(VC_OFF, (unsigned long long)(const void*)vp, DVS, CHUNK,
                      (unsigned long long)NH * DVc);
    }
    if (tid < DKc) {
      const int c = tid;
      float run = 0.f;
      for (int i = 0; i < CHUNK; i++) {
        run += g[(((long)b * Tc + t0 + i) * NH + h) * DKc + c];
        gcf[i * DKc + c] = run;
      }
      glast[c] = run;
    }
    if (tid >= 128 && tid < 128 + CHUNK)
      betac[tid - 128] = beta[((long)b * Tc + t0 + (tid - 128)) * NH + h];
    if (tid == 0) __builtin_amdgcn_s_wait_tensorcnt(0);
    __syncthreads();

    // ---- Phase B: gated tiles kt/km/qt + kmT ----
    for (int e = tid; e < CHUNK * DKc; e += 256) {
      const int i = e >> 7, c = e & 127;
      const float gv = gcf[e];
      const long base = (((long)b * Tc + t0 + i) * NH + h) * DKc + c;
      const float kf = k[base], qf = q[base];
      const float ep = __expf(gv);
      const float em = __expf(fminf(-gv, 80.f));  // clamp vs overflow
      const unsigned short kme = f2bf(kf * em);
      kt[i * 136 + c]  = f2bf(kf * ep);
      km[i * 136 + c]  = kme;
      qt[i * 136 + c]  = f2bf(qf * ep * qscale);
      kmT[c * 72 + i]  = kme;
    }
    __syncthreads();

    // ---- Phase C/D/F: three WMMA matmuls (independent) ----
    {  // C: rks = vc - kt @ S    (64x32 out; 1 tile/wave; K=128)
      const int tM = wave >> 1, tN = wave & 1;
      v8f acc = wmma_acc<128>(kt, tM * 16, 136, St16, tN * 16, 136, laneHi,
                              lanelo, v8f_zero());
      const int i0 = tM * 16 + laneHi * 8;
      const int dv = tN * 16 + lanelo;
#pragma unroll
      for (int r = 0; r < 8; r++)
        rks[(i0 + r) * 32 + dv] = vc[(i0 + r) * 32 + dv] - acc[r];
    }
#pragma unroll
    for (int t = wave; t < 16; t += 8) {  // D: Amat = beta_i*stril(kt@km^T)
      const int tM = t >> 2, tN = t & 3;
      v8f acc = wmma_acc<128>(kt, tM * 16, 136, km, tN * 16, 136, laneHi,
                              lanelo, v8f_zero());
      const int i0 = tM * 16 + laneHi * 8;
      const int j  = tN * 16 + lanelo;
#pragma unroll
      for (int r = 0; r < 8; r++) {
        const int i = i0 + r;
        Amat[i * 64 + j] = (j < i) ? acc[r] * betac[i] : 0.f;
      }
    }
#pragma unroll
    for (int t = wave; t < 16; t += 8) {  // F: At16 = tril_incl(qt@km^T)
      const int tM = t >> 2, tN = t & 3;
      v8f acc = wmma_acc<128>(qt, tM * 16, 136, km, tN * 16, 136, laneHi,
                              lanelo, v8f_zero());
      const int i0 = tM * 16 + laneHi * 8;
      const int j  = tN * 16 + lanelo;
#pragma unroll
      for (int r = 0; r < 8; r++) {
        const int i = i0 + r;
        At16[i * 72 + j] = f2bf((j <= i) ? acc[r] : 0.f);
      }
    }
    __syncthreads();

    // ---- Phase E: forward substitution (I + Amat) v_new = beta*rks ----
    if (tid < DVS) {
      const int dv = tid;
      float* vrow = vntf + dv * 64;
      for (int i = 0; i < CHUNK; i++) {
        float acc = betac[i] * rks[i * 32 + dv];
        for (int j = 0; j < i; j++) acc -= Amat[i * 64 + j] * vrow[j];
        vrow[i] = acc;
        vnt16[dv * 72 + i] = f2bf(acc);
      }
    }
    __syncthreads();

    // ---- Phase G: o = qt @ S + attn @ v_new   (64x32 out; 1 tile/wave) ----
    {
      const int tM = wave >> 1, tN = wave & 1;
      v8f acc = wmma_acc<128>(qt, tM * 16, 136, St16, tN * 16, 136, laneHi,
                              lanelo, v8f_zero());
      acc = wmma_acc<64>(At16, tM * 16, 72, vnt16, tN * 16, 72, laneHi, lanelo,
                         acc);
      const int i0 = tM * 16 + laneHi * 8;
      const int dv = tN * 16 + lanelo;
#pragma unroll
      for (int r = 0; r < 8; r++)
        o[(((long)b * Tc + t0 + i0 + r) * NH + h) * DVc + vs * DVS + dv] =
            acc[r];
    }
    __syncthreads();

    // ---- Phase H: S^T = e^{g_last[d]} * (S^T + v_new^T @ km)  (32x128) ----
#pragma unroll
    for (int t = wave; t < 16; t += 8) {
      const int tM = t >> 3, tN = t & 7;
      v8f acc = wmma_acc<64>(vnt16, tM * 16, 72, kmT, tN * 16, 72, laneHi,
                             lanelo, v8f_zero());
      const int dv0 = tM * 16 + laneHi * 8;
      const int d   = tN * 16 + lanelo;
      const float ge = __expf(glast[d]);
#pragma unroll
      for (int r = 0; r < 8; r++) {
        const int dv = dv0 + r;
        float s = Stf[dv * 128 + d];
        s = ge * (s + acc[r]);
        Stf[dv * 128 + d] = s;
        St16[dv * 136 + d] = f2bf(s);
      }
    }
    __syncthreads();
  }
}

// ---------------------------------------------------------------------------
// RMS norm over DV + sigmoid gate -> bf16
// ---------------------------------------------------------------------------
__global__ __launch_bounds__(128) void gate_rms_kernel(
    const float* __restrict__ o, const float* __restrict__ gout,
    const float* __restrict__ rms_scale, unsigned short* __restrict__ og16) {
  __shared__ float red[128];
  const long row = blockIdx.x;
  const int  d   = threadIdx.x;
  const long base = row * DVc + d;
  const float x = o[base];
  red[d] = x * x;
  __syncthreads();
  for (int s = 64; s > 0; s >>= 1) {
    if (d < s) red[d] += red[d + s];
    __syncthreads();
  }
  const float r  = rsqrtf(red[0] / (float)DVc + EPSf);
  const float gv = gout[base];
  og16[base] = f2bf(x * r * rms_scale[d] * (1.f / (1.f + __expf(-gv))));
}

// ---------------------------------------------------------------------------
// Host orchestration
// ---------------------------------------------------------------------------
static inline void launch_gemm(const unsigned short* A, const unsigned short* BT,
                               float* C, int Md, int Nd, int Kd, hipStream_t s) {
  dim3 grid((Nd + 127) / 128, (Md + 127) / 128);
  gemm_bf16_wmma<<<grid, 256, 0, s>>>(A, BT, C, Md, Nd, Kd);
}
static inline void launch_cast(const float* src, unsigned short* dst, long n,
                               hipStream_t s) {
  cast_bf16_kernel<<<2048, 256, 0, s>>>(src, dst, n);
}
static inline void launch_castT(const float* src, unsigned short* dst, int Kd,
                                int Nd, hipStream_t s) {
  dim3 grid((Kd + 31) / 32, (Nd + 31) / 32);
  castT_bf16_kernel<<<grid, 256, 0, s>>>(src, dst, Kd, Nd);
}

extern "C" void kernel_launch(void* const* d_in, const int* in_sizes, int n_in,
                              void* d_out, int out_size, void* d_ws,
                              size_t ws_size, hipStream_t stream) {
  const float* x     = (const float*)d_in[0];
  const float* Wq    = (const float*)d_in[1];
  const float* Wk    = (const float*)d_in[2];
  const float* Wv    = (const float*)d_in[3];
  const float* convq = (const float*)d_in[4];
  const float* convk = (const float*)d_in[5];
  const float* convv = (const float*)d_in[6];
  const float* Wf1   = (const float*)d_in[7];
  const float* Wf2   = (const float*)d_in[8];
  const float* Wb    = (const float*)d_in[9];
  const float* A_log = (const float*)d_in[10];
  const float* dtb   = (const float*)d_in[11];
  const float* Wg1   = (const float*)d_in[12];
  const float* Wg2   = (const float*)d_in[13];
  const float* rmss  = (const float*)d_in[14];
  const float* Wo    = (const float*)d_in[15];

  char* ws = (char*)d_ws;
  size_t off = 0;
  auto alloc = [&](size_t bytes) -> char* {
    char* p = ws + off;
    off += (bytes + 255) & ~(size_t)255;
    return p;
  };

  unsigned short* x16   = (unsigned short*)alloc((size_t)Mrows * HID * 2);
  unsigned short* wq16  = (unsigned short*)alloc((size_t)HID * KEY_DIM * 2);
  unsigned short* wk16  = (unsigned short*)alloc((size_t)HID * KEY_DIM * 2);
  unsigned short* wv16  = (unsigned short*)alloc((size_t)HID * VAL_DIM * 2);
  unsigned short* wo16  = (unsigned short*)alloc((size_t)VAL_DIM * HID * 2);
  unsigned short* wf116 = (unsigned short*)alloc((size_t)HID * DVc * 2);
  unsigned short* wf216 = (unsigned short*)alloc((size_t)DVc * KEY_DIM * 2);
  unsigned short* wg116 = (unsigned short*)alloc((size_t)HID * DVc * 2);
  unsigned short* wg216 = (unsigned short*)alloc((size_t)DVc * VAL_DIM * 2);
  unsigned short* wb16  = (unsigned short*)alloc((size_t)HID * NH * 2);
  float*          betab = (float*)alloc((size_t)Mrows * NH * 4);
  float*          f1buf = (float*)alloc((size_t)Mrows * DVc * 4);
  unsigned short* f116  = (unsigned short*)alloc((size_t)Mrows * DVc * 2);
  float*          g1buf = (float*)alloc((size_t)Mrows * DVc * 4);
  unsigned short* g116  = (unsigned short*)alloc((size_t)Mrows * DVc * 2);
  float*          P0    = (float*)alloc((size_t)Mrows * KEY_DIM * 4);
  float*          P1    = (float*)alloc((size_t)Mrows * KEY_DIM * 4);
  float*          P2    = (float*)alloc((size_t)Mrows * VAL_DIM * 4);
  float*          Qb    = (float*)alloc((size_t)Mrows * KEY_DIM * 4);
  float*          Kb    = (float*)alloc((size_t)Mrows * KEY_DIM * 4);
  float*          Vb    = (float*)alloc((size_t)Mrows * VAL_DIM * 4);
  unsigned short* og16  = (unsigned short*)alloc((size_t)Mrows * VAL_DIM * 2);
  (void)ws_size; (void)in_sizes; (void)n_in; (void)out_size;

  // 1) casts: activations row-major, weights transposed (N x K)
  launch_cast(x, x16, (long)Mrows * HID, stream);
  launch_castT(Wq,  wq16,  HID, KEY_DIM, stream);
  launch_castT(Wk,  wk16,  HID, KEY_DIM, stream);
  launch_castT(Wv,  wv16,  HID, VAL_DIM, stream);
  launch_castT(Wo,  wo16,  VAL_DIM, HID, stream);
  launch_castT(Wf1, wf116, HID, DVc,     stream);
  launch_castT(Wf2, wf216, DVc, KEY_DIM, stream);
  launch_castT(Wg1, wg116, HID, DVc,     stream);
  launch_castT(Wg2, wg216, DVc, VAL_DIM, stream);
  launch_castT(Wb,  wb16,  HID, NH,      stream);

  // 2) projections (WMMA)
  launch_gemm(x16, wq16, P0,    Mrows, KEY_DIM, HID, stream);
  launch_gemm(x16, wk16, P1,    Mrows, KEY_DIM, HID, stream);
  launch_gemm(x16, wv16, P2,    Mrows, VAL_DIM, HID, stream);
  launch_gemm(x16, wb16, betab, Mrows, NH,      HID, stream);
  sigmoid_kernel<<<512, 256, 0, stream>>>(betab, (long)Mrows * NH);

  // 3) causal depthwise conv + SiLU
  conv_silu_kernel<<<4096, 256, 0, stream>>>(P0, convq, Qb, KEY_DIM);
  conv_silu_kernel<<<4096, 256, 0, stream>>>(P1, convk, Kb, KEY_DIM);
  conv_silu_kernel<<<4096, 256, 0, stream>>>(P2, convv, Vb, VAL_DIM);

  // 4) g = -exp(A_log) * softplus(x@Wf1@Wf2 + dt_bias)  (into P0)
  launch_gemm(x16, wf116, f1buf, Mrows, DVc, HID, stream);
  launch_cast(f1buf, f116, (long)Mrows * DVc, stream);
  launch_gemm(f116, wf216, P0, Mrows, KEY_DIM, DVc, stream);
  f_to_g_kernel<<<4096, 256, 0, stream>>>(P0, A_log, dtb);

  // 5) chunked delta-rule attention (WMMA + TDM) -> o in P2
  kda_chunk_kernel<<<Bc * NH * NSPLIT, 256, ATT_SMEM, stream>>>(
      Qb, Kb, Vb, P0, betab, P2);

  // 6) output gate g_out = x@Wg1@Wg2  (into P1)
  launch_gemm(x16, wg116, g1buf, Mrows, DVc, HID, stream);
  launch_cast(g1buf, g116, (long)Mrows * DVc, stream);
  launch_gemm(g116, wg216, P1, Mrows, VAL_DIM, DVc, stream);

  // 7) RMS norm + sigmoid gate -> bf16
  gate_rms_kernel<<<Mrows * NH, 128, 0, stream>>>(P2, P1, rmss, og16);

  // 8) final projection -> d_out (f32)
  launch_gemm(og16, wo16, (float*)d_out, Mrows, HID, VAL_DIM, stream);
}
